// DGCNNCls_37417755082769
// MI455X (gfx1250) — matrix operand order repaired and checked by reading, not compiled, not run on hardware
//
#include <hip/hip_runtime.h>
#include <hip/hip_bf16.h>

// ---------------------------------------------------------------------------
// DGCNN forward for MI455X (gfx1250, wave32, WMMA).
// All GEMM-shaped work (KNN inner products, EdgeConv 1x1 convs, conv5) uses
// v_wmma_f32_16x16x32_bf16 with a rolling 2-deep B-fragment pipeline: the
// next fragment's load is issued before the WMMA consuming the current one,
// so loads overlap the XDL pipe with only 2 fragments live. Epilogues
// (BN + leaky, neighbor-max, global max/mean pooling) are fused.
// ---------------------------------------------------------------------------

typedef __attribute__((ext_vector_type(16))) __bf16 v16bf;
typedef __attribute__((ext_vector_type(8)))  float  v8f;

#define BATCH 16
#define NPTS  2048
#define KNN_K 20

typedef unsigned short us16;

// ---- bf16 helpers (storage = raw 16-bit payload) --------------------------
__device__ __forceinline__ float bload(const us16* p) {
  return __uint_as_float(((unsigned)(*p)) << 16);
}
__device__ __forceinline__ us16 bpack(float f) {
  unsigned u = __float_as_uint(f);
  u += 0x7FFFu + ((u >> 16) & 1u);            // round-to-nearest-even
  return (us16)(u >> 16);
}
__device__ __forceinline__ __bf16 bfbits(us16 u) {
  return __builtin_bit_cast(__bf16, u);
}
// ---- ordered-float <-> uint for atomic max --------------------------------
__device__ __forceinline__ unsigned encf(float f) {
  unsigned u = __float_as_uint(f);
  return u ^ ((unsigned)((int)u >> 31) | 0x80000000u);
}
__device__ __forceinline__ float decf(unsigned e) {
  unsigned bits = (e & 0x80000000u) ? (e ^ 0x80000000u) : ~e;
  return __uint_as_float(bits);
}
// ---- fragment loaders (contiguous K, same order for A and B) --------------------------------
// A (16x32 bf16): lane holds row (lane&15); 16 contiguous K at
// base = ks*32 + hi*16 (matches B).
__device__ __forceinline__ v16bf loadA16(const us16* __restrict__ row, int base) {
  v16bf a;
#pragma unroll
  for (int e = 0; e < 8; ++e) a[e]     = bfbits(row[base + e]);
#pragma unroll
  for (int e = 0; e < 8; ++e) a[8 + e] = bfbits(row[base + 8 + e]);
  return a;
}
// B (32x16 bf16): lane holds column (lane&15); 16 contiguous K at kb = ks*32+hi*16.
__device__ __forceinline__ v16bf loadB16(const us16* __restrict__ row, int kb) {
  v16bf b;
#pragma unroll
  for (int e = 0; e < 16; ++e) b[e] = bfbits(row[kb + e]);
  return b;
}

// ===========================================================================
// KNN: one 16-query tile per workgroup vs all 2048 candidates.
// NK = Cpad/32 k-steps. dist tile lives in dynamic LDS (16 x 2048 f32 = 128KB).
// B fragments rotate across (candidate-tile x k-step).
// ===========================================================================
template <int NK>
__global__ __launch_bounds__(256) void knn_kernel(
    const us16* __restrict__ xb, int stride, int colOff,
    const float* __restrict__ norms, int* __restrict__ idx)
{
  extern __shared__ float dist[];              // [16][NPTS]
  const int b    = blockIdx.y;
  const int q0   = blockIdx.x * 16;
  const int tid  = threadIdx.x;
  const int lane = tid & 31;
  const int wave = tid >> 5;
  const int col  = lane & 15;
  const int hi   = lane >> 4;

  const us16* xbB = xb + (size_t)b * NPTS * stride + colOff;

  // Preload A fragments (query rows) once.
  v16bf afrag[NK];
  {
    const us16* qrow = xbB + (size_t)(q0 + col) * stride;
#pragma unroll
    for (int ks = 0; ks < NK; ++ks)
      afrag[ks] = loadA16(qrow, ks * 32 + hi * 16);
  }
  float qnr[8];
#pragma unroll
  for (int r = 0; r < 8; ++r)
    qnr[r] = norms[b * NPTS + q0 + r + hi * 8];

  // Rolling B pipeline over candidate tiles.
  v16bf bcur = loadB16(xbB + (size_t)(wave * 16 + col) * stride, hi * 16);
#pragma unroll 1
  for (int t = wave; t < NPTS / 16; t += 8) {
    const us16* crow  = xbB + (size_t)(t * 16 + col) * stride;
    const int   tn    = (t + 8 < NPTS / 16) ? t + 8 : t;     // clamp (no OOB)
    const us16* crown = xbB + (size_t)(tn * 16 + col) * stride;

    v8f acc = {};
#pragma unroll
    for (int ks = 0; ks < NK; ++ks) {
      v16bf bnext = (ks + 1 < NK) ? loadB16(crow, (ks + 1) * 32 + hi * 16)
                                  : loadB16(crown, hi * 16);
      acc = __builtin_amdgcn_wmma_f32_16x16x32_bf16(
          false, afrag[ks], false, bcur, (short)0, acc, false, false);
      bcur = bnext;
    }
    const float cn = norms[b * NPTS + t * 16 + col];
#pragma unroll
    for (int r = 0; r < 8; ++r) {
      const int m = r + hi * 8;
      dist[m * NPTS + t * 16 + col] = 2.0f * acc[r] - qnr[r] - cn;
    }
  }
  __syncthreads();

  // Top-20 per row: wave handles rows {wave, wave+8}; 20 argmax rounds.
  volatile float* vd = dist;
  for (int rr = 0; rr < 2; ++rr) {
    const int m = wave + rr * 8;
    for (int kk = 0; kk < KNN_K; ++kk) {
      float bv = -3.0e38f; int bi = 0x7FFFFFFF;
      for (int j = lane; j < NPTS; j += 32) {
        const float v = vd[m * NPTS + j];
        if (v > bv || (v == bv && j < bi)) { bv = v; bi = j; }
      }
#pragma unroll
      for (int off = 16; off > 0; off >>= 1) {
        const float ov = __shfl_xor(bv, off, 32);
        const int   oi = __shfl_xor(bi, off, 32);
        if (ov > bv || (ov == bv && oi < bi)) { bv = ov; bi = oi; }
      }
      if (lane == 0) {
        idx[((size_t)b * NPTS + q0 + m) * KNN_K + kk] = bi;
        vd[m * NPTS + bi] = -3.4e38f;
      }
    }
  }
}

// ===========================================================================
// EdgeConv: workgroup = 16 points (320 edges -> 20 M-tiles) x all O channels.
// A fragments (gathered edge features) built ONCE per M-tile, cached in
// registers; B (weights) rotate across (output-tile x k-step).
// f[e][c] = (c<C ? x[nbr][c]-x[p][c] : (c<2C ? x[p][c-C] : 0)); h = f * W^T;
// BN + leaky; max over 20 neighbors via ordered-uint LDS atomics. NK=K2pad/32.
// ===========================================================================
__device__ __forceinline__ __bf16 edgeval(const us16* ctr, const us16* nbr,
                                          int k, int Cin, int twoC) {
  if (k < Cin)  return bfbits(bpack(bload(nbr + k) - bload(ctr + k)));
  if (k < twoC) return bfbits(ctr[k - Cin]);
  return bfbits((us16)0);
}

template <int NK>
__global__ __launch_bounds__(256) void edgeconv_kernel(
    const us16* __restrict__ xb, int strideIn, int colOffIn, int Cin,
    const int* __restrict__ idx,
    const us16* __restrict__ Wb,
    const float* __restrict__ bng, const float* __restrict__ bnb,
    const float* __restrict__ bnm, const float* __restrict__ bnv,
    us16* __restrict__ out, int strideOut, int colOffOut, int O)
{
  __shared__ int      sidx[16 * KNN_K];
  __shared__ unsigned smax[16 * 256];
  const int b   = blockIdx.y;
  const int p0  = blockIdx.x * 16;
  const int tid = threadIdx.x;

  for (int i = tid; i < 16 * KNN_K; i += 256)
    sidx[i] = idx[((size_t)b * NPTS + p0) * KNN_K + i];
  for (int i = tid; i < 16 * O; i += 256)
    smax[i] = 0u;                              // below every real encoding
  __syncthreads();

  const int lane = tid & 31, wave = tid >> 5;
  const int col  = lane & 15, hi = lane >> 4;
  const int twoC  = 2 * Cin;
  const int K2pad = NK * 32;
  const int nN    = O >> 4;
  const us16* xbB = xb + (size_t)b * NPTS * strideIn + colOffIn;

  for (int mt = wave; mt < 20; mt += 8) {
    // prefetch next M-tile's gather rows (rotation can't cover indexed rows)
    if (mt + 8 < 20) {
      const int le2 = (mt + 8) * 16 + col;
      const int pl2 = le2 / KNN_K, j2 = le2 % KNN_K;
      __builtin_prefetch(xbB + (size_t)(p0 + pl2) * strideIn, 0, 0);
      __builtin_prefetch(xbB + (size_t)sidx[pl2 * KNN_K + j2] * strideIn, 0, 0);
    }
    // ---- build A fragments once for this M-tile ----
    const int le = mt * 16 + col;
    const int pl = le / KNN_K, j = le % KNN_K;
    const us16* ctr = xbB + (size_t)(p0 + pl) * strideIn;
    const us16* nbr = xbB + (size_t)sidx[pl * KNN_K + j] * strideIn;
    v16bf afrag[NK];
#pragma unroll
    for (int ks = 0; ks < NK; ++ks) {
      v16bf a;
      const int base = ks * 32 + hi * 16;
#pragma unroll
      for (int e = 0; e < 8; ++e) a[e]     = edgeval(ctr, nbr, base + e, Cin, twoC);
#pragma unroll
      for (int e = 0; e < 8; ++e) a[8 + e] = edgeval(ctr, nbr, base + 8 + e, Cin, twoC);
      afrag[ks] = a;
    }
    // ---- sweep output-channel tiles with rolling B pipeline ----
    v16bf bcur = loadB16(Wb + (size_t)col * K2pad, hi * 16);
#pragma unroll 1
    for (int nt = 0; nt < nN; ++nt) {
      const int oc = nt * 16 + col;
      const us16* wrow  = Wb + (size_t)oc * K2pad;
      const us16* wnext = Wb + (size_t)((nt + 1 < nN ? oc + 16 : col)) * K2pad;
      v8f acc = {};
#pragma unroll
      for (int ks = 0; ks < NK; ++ks) {
        v16bf bnext = (ks + 1 < NK) ? loadB16(wrow, (ks + 1) * 32 + hi * 16)
                                    : loadB16(wnext, hi * 16);
        acc = __builtin_amdgcn_wmma_f32_16x16x32_bf16(
            false, afrag[ks], false, bcur, (short)0, acc, false, false);
        bcur = bnext;
      }
      // BN + leaky + neighbor-max
      const float sc = bng[oc] * rsqrtf(bnv[oc] + 1e-5f);
      const float sh = bnb[oc] - sc * bnm[oc];
#pragma unroll
      for (int r = 0; r < 8; ++r) {
        const int m   = r + hi * 8;
        const int pl2 = (mt * 16 + m) / KNN_K;
        float h = sc * acc[r] + sh;
        h = (h >= 0.f) ? h : 0.2f * h;
        atomicMax(&smax[pl2 * O + oc], encf(h));
      }
    }
  }
  __syncthreads();

  us16* outB = out + (size_t)b * NPTS * strideOut + colOffOut;
  for (int i = tid; i < 16 * O; i += 256) {
    const int pl = i / O, oc = i % O;
    outB[(size_t)(p0 + pl) * strideOut + oc] = bpack(decf(smax[i]));
  }
}

// ===========================================================================
// conv5 (512 -> 1024) + BN + leaky + fused global max/mean pooling.
// Workgroup: 16 points; waves each own 8 output-channel tiles; K = 512.
// A rotates across k-steps; B rotates across (k-step x 8 tiles).
// ===========================================================================
__global__ __launch_bounds__(256) void conv5_kernel(
    const us16* __restrict__ xcat, const us16* __restrict__ W5b,
    const float* __restrict__ bng, const float* __restrict__ bnb,
    const float* __restrict__ bnm, const float* __restrict__ bnv,
    unsigned* __restrict__ gmax, float* __restrict__ gsum)
{
  const int b   = blockIdx.y;
  const int p0  = blockIdx.x * 16;
  const int tid = threadIdx.x;
  const int lane = tid & 31, wave = tid >> 5;
  const int col  = lane & 15, hi = lane >> 4;

  const us16* prow = xcat + ((size_t)b * NPTS + p0 + col) * 512;
  const us16* wbase = W5b + (size_t)(wave * 128 + col) * 512;  // tile t row: +t*16*512

  v8f acc[8];
  v8f z = {};
#pragma unroll
  for (int t = 0; t < 8; ++t) acc[t] = z;

  v16bf acur = loadA16(prow, hi * 16);
  v16bf bcur = loadB16(wbase, hi * 16);

#pragma unroll 1
  for (int ks = 0; ks < 15; ++ks) {            // K = 512, 16 k-steps (last peeled)
    v16bf anext = loadA16(prow, (ks + 1) * 32 + hi * 16);
#pragma unroll
    for (int t = 0; t < 8; ++t) {
      v16bf bnext = (t < 7)
          ? loadB16(wbase + (size_t)(t + 1) * 16 * 512, ks * 32 + hi * 16)
          : loadB16(wbase, (ks + 1) * 32 + hi * 16);
      acc[t] = __builtin_amdgcn_wmma_f32_16x16x32_bf16(
          false, acur, false, bcur, (short)0, acc[t], false, false);
      bcur = bnext;
    }
    acur = anext;
  }
#pragma unroll
  for (int t = 0; t < 8; ++t) {                // ks = 15
    v16bf bnext = (t < 7)
        ? loadB16(wbase + (size_t)(t + 1) * 16 * 512, 15 * 32 + hi * 16)
        : bcur;
    acc[t] = __builtin_amdgcn_wmma_f32_16x16x32_bf16(
        false, acur, false, bcur, (short)0, acc[t], false, false);
    bcur = bnext;
  }

#pragma unroll
  for (int t = 0; t < 8; ++t) {
    const int oc = (wave * 8 + t) * 16 + col;
    const float sc = bng[oc] * rsqrtf(bnv[oc] + 1e-5f);
    const float sh = bnb[oc] - sc * bnm[oc];
    float mx = -3.4e38f, sm = 0.f;
#pragma unroll
    for (int r = 0; r < 8; ++r) {
      float h = sc * acc[t][r] + sh;
      h = (h >= 0.f) ? h : 0.2f * h;
      mx = fmaxf(mx, h);
      sm += h;
    }
    mx = fmaxf(mx, __shfl_xor(mx, 16, 32));    // combine the two M-halves
    sm += __shfl_xor(sm, 16, 32);
    if (hi == 0) {
      atomicMax(&gmax[b * 1024 + oc], encf(mx));
      atomicAdd(&gsum[b * 1024 + oc], sm);
    }
  }
}

// ===========================================================================
// Small utility kernels
// ===========================================================================
__global__ __launch_bounds__(256) void prep_input_kernel(
    const float* __restrict__ x, us16* __restrict__ xb0)
{
  const int i = blockIdx.x * 256 + threadIdx.x;     // point id
  if (i >= BATCH * NPTS) return;
  const int b = i / NPTS, n = i % NPTS;
  for (int c = 0; c < 32; ++c) {
    const float v = (c < 3) ? x[((size_t)b * 3 + c) * NPTS + n] : 0.f;
    xb0[(size_t)i * 32 + c] = bpack(v);
  }
}

__global__ __launch_bounds__(256) void wconv_kernel(
    const float* __restrict__ src, us16* __restrict__ dst,
    int O, int Kin, int Kpad)
{
  const int i = blockIdx.x * 256 + threadIdx.x;
  if (i >= O * Kpad) return;
  const int o = i / Kpad, k = i % Kpad;
  dst[i] = bpack((k < Kin) ? src[(size_t)o * Kin + k] : 0.f);
}

__global__ __launch_bounds__(256) void norms_kernel(
    const us16* __restrict__ xb, int stride, int colOff, int C,
    float* __restrict__ norms)
{
  const int i = blockIdx.x * 256 + threadIdx.x;
  if (i >= BATCH * NPTS) return;
  const us16* row = xb + (size_t)i * stride + colOff;
  float s = 0.f;
  for (int c = 0; c < C; ++c) { const float f = bload(row + c); s += f * f; }
  norms[i] = s;
}

__global__ __launch_bounds__(256) void initpool_kernel(
    unsigned* __restrict__ gmax, float* __restrict__ gsum)
{
  const int i = blockIdx.x * 256 + threadIdx.x;
  if (i >= BATCH * 1024) return;
  gmax[i] = 0u;
  gsum[i] = 0.f;
}

__global__ __launch_bounds__(256) void poolfin_kernel(
    const unsigned* __restrict__ gmax, const float* __restrict__ gsum,
    float* __restrict__ g)
{
  const int i = blockIdx.x * 256 + threadIdx.x;
  if (i >= BATCH * 1024) return;
  const int b = i / 1024, c = i % 1024;
  g[(size_t)b * 2048 + c]        = decf(gmax[i]);
  g[(size_t)b * 2048 + 1024 + c] = gsum[i] * (1.f / (float)NPTS);
}

__global__ __launch_bounds__(256) void lin_kernel(
    const float* __restrict__ in, const float* __restrict__ W,
    const float* __restrict__ bias,
    const float* __restrict__ gw, const float* __restrict__ gb,
    const float* __restrict__ gm, const float* __restrict__ gv,
    float* __restrict__ out, int I, int O, int act)
{
  const int i = blockIdx.x * 256 + threadIdx.x;
  if (i >= BATCH * O) return;
  const int b = i / O, o = i % O;
  const float* row = in + (size_t)b * I;
  const float* w   = W + (size_t)o * I;
  float s = 0.f;
  for (int k = 0; k < I; ++k) s += row[k] * w[k];
  if (bias) s += bias[o];
  if (gw) s = gw[o] * (s - gm[o]) * rsqrtf(gv[o] + 1e-5f) + gb[o];
  if (act) s = (s >= 0.f) ? s : 0.2f * s;
  out[i] = s;
}

// ===========================================================================
// Orchestration
// ===========================================================================
extern "C" void kernel_launch(void* const* d_in, const int* in_sizes, int n_in,
                              void* d_out, int out_size, void* d_ws, size_t ws_size,
                              hipStream_t stream) {
  (void)in_sizes; (void)n_in; (void)out_size; (void)ws_size;
  // Input flattening: x, then params leaves in dict order, BN tuples expanded.
  const float* x   = (const float*)d_in[0];
  const float* w1  = (const float*)d_in[1];
  const float* bn1g = (const float*)d_in[2],  *bn1b = (const float*)d_in[3],
             * bn1m = (const float*)d_in[4],  *bn1v = (const float*)d_in[5];
  const float* w2  = (const float*)d_in[6];
  const float* bn2g = (const float*)d_in[7],  *bn2b = (const float*)d_in[8],
             * bn2m = (const float*)d_in[9],  *bn2v = (const float*)d_in[10];
  const float* w3  = (const float*)d_in[11];
  const float* bn3g = (const float*)d_in[12], *bn3b = (const float*)d_in[13],
             * bn3m = (const float*)d_in[14], *bn3v = (const float*)d_in[15];
  const float* w4  = (const float*)d_in[16];
  const float* bn4g = (const float*)d_in[17], *bn4b = (const float*)d_in[18],
             * bn4m = (const float*)d_in[19], *bn4v = (const float*)d_in[20];
  const float* w5  = (const float*)d_in[21];
  const float* bn5g = (const float*)d_in[22], *bn5b = (const float*)d_in[23],
             * bn5m = (const float*)d_in[24], *bn5v = (const float*)d_in[25];
  const float* l1w = (const float*)d_in[26];
  const float* bn6g = (const float*)d_in[27], *bn6b = (const float*)d_in[28],
             * bn6m = (const float*)d_in[29], *bn6v = (const float*)d_in[30];
  const float* l2w = (const float*)d_in[31];
  const float* l2b = (const float*)d_in[32];
  const float* bn7g = (const float*)d_in[33], *bn7b = (const float*)d_in[34],
             * bn7m = (const float*)d_in[35], *bn7v = (const float*)d_in[36];
  const float* l3w = (const float*)d_in[37];
  const float* l3b = (const float*)d_in[38];

  char* ws = (char*)d_ws;
  us16*     xb0   = (us16*)(ws + 0);                 //  2,097,152  (B,N,32) bf16
  us16*     xcat  = (us16*)(ws + 2097152);           // 33,554,432  (B,N,512) bf16
  float*    norms = (float*)(ws + 35651584);         //    131,072
  int*      idx   = (int*)(ws + 35782656);           //  2,621,440
  us16*     W1b   = (us16*)(ws + 38404096);          //      4,096  64x32
  us16*     W2b   = (us16*)(ws + 38408192);          //     16,384  64x128
  us16*     W3b   = (us16*)(ws + 38424576);          //     32,768  128x128
  us16*     W4b   = (us16*)(ws + 38457344);          //    131,072  256x256
  us16*     W5b   = (us16*)(ws + 38588416);          //  1,048,576  1024x512
  unsigned* gmax  = (unsigned*)(ws + 39636992);      //     65,536
  float*    gsum  = (float*)(ws + 39702528);         //     65,536
  float*    g     = (float*)(ws + 39768064);         //    131,072  (B,2048)
  float*    h1    = (float*)(ws + 39899136);         //     32,768  (B,512)
  float*    h2    = (float*)(ws + 39931904);         //     16,384  (B,256)
  float*    out   = (float*)d_out;                   //  (B,40)

  const dim3 blk(256);
  const dim3 gridPts(BATCH * NPTS / 256);            // 128
  const dim3 gridKnn(NPTS / 16, BATCH);              // 128 x 16
  const dim3 gridEdge(NPTS / 16, BATCH);             // 128 x 16
  const size_t knnLds = 16 * NPTS * sizeof(float);   // 128 KB (CDNA5: 320KB/WG)

  // ---- prep: input transpose/pad + weight conversion ----
  prep_input_kernel<<<gridPts, blk, 0, stream>>>(x, xb0);
  wconv_kernel<<<(64 * 32 + 255) / 256, blk, 0, stream>>>(w1, W1b, 64, 6, 32);
  wconv_kernel<<<(64 * 128 + 255) / 256, blk, 0, stream>>>(w2, W2b, 64, 128, 128);
  wconv_kernel<<<(128 * 128 + 255) / 256, blk, 0, stream>>>(w3, W3b, 128, 128, 128);
  wconv_kernel<<<(256 * 256 + 255) / 256, blk, 0, stream>>>(w4, W4b, 256, 256, 256);
  wconv_kernel<<<(1024 * 512 + 255) / 256, blk, 0, stream>>>(w5, W5b, 1024, 512, 512);

  // ---- layer 1: C=3 (padded 32), O=64 -> xcat[:, 0:64] ----
  norms_kernel<<<gridPts, blk, 0, stream>>>(xb0, 32, 0, 32, norms);
  knn_kernel<1><<<gridKnn, blk, knnLds, stream>>>(xb0, 32, 0, norms, idx);
  edgeconv_kernel<1><<<gridEdge, blk, 0, stream>>>(
      xb0, 32, 0, 3, idx, W1b, bn1g, bn1b, bn1m, bn1v, xcat, 512, 0, 64);

  // ---- layer 2: C=64, O=64 -> xcat[:, 64:128] ----
  norms_kernel<<<gridPts, blk, 0, stream>>>(xcat, 512, 0, 64, norms);
  knn_kernel<2><<<gridKnn, blk, knnLds, stream>>>(xcat, 512, 0, norms, idx);
  edgeconv_kernel<4><<<gridEdge, blk, 0, stream>>>(
      xcat, 512, 0, 64, idx, W2b, bn2g, bn2b, bn2m, bn2v, xcat, 512, 64, 64);

  // ---- layer 3: C=64, O=128 -> xcat[:, 128:256] ----
  norms_kernel<<<gridPts, blk, 0, stream>>>(xcat, 512, 64, 64, norms);
  knn_kernel<2><<<gridKnn, blk, knnLds, stream>>>(xcat, 512, 64, norms, idx);
  edgeconv_kernel<4><<<gridEdge, blk, 0, stream>>>(
      xcat, 512, 64, 64, idx, W3b, bn3g, bn3b, bn3m, bn3v, xcat, 512, 128, 128);

  // ---- layer 4: C=128, O=256 -> xcat[:, 256:512] ----
  norms_kernel<<<gridPts, blk, 0, stream>>>(xcat, 512, 128, 128, norms);
  knn_kernel<4><<<gridKnn, blk, knnLds, stream>>>(xcat, 512, 128, norms, idx);
  edgeconv_kernel<8><<<gridEdge, blk, 0, stream>>>(
      xcat, 512, 128, 128, idx, W4b, bn4g, bn4b, bn4m, bn4v, xcat, 512, 256, 256);

  // ---- conv5 + fused global max/mean pooling ----
  initpool_kernel<<<(BATCH * 1024 + 255) / 256, blk, 0, stream>>>(gmax, gsum);
  conv5_kernel<<<dim3(NPTS / 16, BATCH), blk, 0, stream>>>(
      xcat, W5b, bn5g, bn5b, bn5m, bn5v, gmax, gsum);
  poolfin_kernel<<<(BATCH * 1024 + 255) / 256, blk, 0, stream>>>(gmax, gsum, g);

  // ---- MLP head (tiny; scalar) ----
  lin_kernel<<<(BATCH * 512 + 255) / 256, blk, 0, stream>>>(
      g, l1w, nullptr, bn6g, bn6b, bn6m, bn6v, h1, 2048, 512, 1);
  lin_kernel<<<(BATCH * 256 + 255) / 256, blk, 0, stream>>>(
      h1, l2w, l2b, bn7g, bn7b, bn7m, bn7v, h2, 512, 256, 1);
  lin_kernel<<<(BATCH * 40 + 255) / 256, blk, 0, stream>>>(
      h2, l3w, l3b, nullptr, nullptr, nullptr, nullptr, out, 256, 40, 0);
}
